// SplitNetworkAttention_80333068304850
// MI455X (gfx1250) — matrix operand order repaired and checked by reading, not compiled, not run on hardware
//
#include <hip/hip_runtime.h>
#include <hip/hip_bf16.h>
#include <math.h>

// ---------------- problem constants ----------------
#define B_   16384
#define DX_  512
#define DS_  512
#define H1_  256
#define H2_  256
#define O_   6
#define NH_  16
#define A_   128

typedef __attribute__((ext_vector_type(16))) _Float16 v16h;
typedef __attribute__((ext_vector_type(8)))  _Float16 v8h;
typedef __attribute__((ext_vector_type(8)))  float    v8f;

// ---------------- workspace layout (bytes) ----------------
static constexpr size_t OFF_WST  = 0;                                 // 512*512 f16
static constexpr size_t OFF_W1T  = OFF_WST  + (size_t)512*512*2;      // 16*512*256 f16
static constexpr size_t OFF_W2T  = OFF_W1T  + (size_t)16*512*256*2;   // 16*256*256 f16
static constexpr size_t OFF_WOT  = OFF_W2T  + (size_t)16*256*256*2;   // 16*256*16 f16
static constexpr size_t OFF_WQT  = OFF_WOT  + (size_t)16*256*16*2;    // 256*128 f16
static constexpr size_t OFF_X16  = OFF_WQT  + (size_t)256*128*2;      // B*512 f16
static constexpr size_t OFF_HID  = OFF_X16  + (size_t)B_*512*2;       // B*512 f16
static constexpr size_t OFF_Q    = OFF_HID  + (size_t)B_*512*2;       // B*16*128 f16
static constexpr size_t OFF_MEAN = OFF_Q    + (size_t)B_*16*128*2;    // B*16*6 f32
static constexpr size_t OFF_STD  = OFF_MEAN + (size_t)B_*16*6*4;      // B*16*6 f32
static constexpr size_t OFF_S    = OFF_STD  + (size_t)B_*16*6*4;      // 1 f32

// ---------------- fragment helpers ----------------
// A-fragment (16x32 f16): lane<16 -> row=lane, K {k0..k0+7} and {k0+16..k0+23};
// lane>=16 -> same row, K offsets +8. Caller passes per-lane base pointer.
__device__ __forceinline__ v16h frag_gap16(const _Float16* p) {
  v8h lo = *(const v8h*)(p);
  v8h hi = *(const v8h*)(p + 16);
  v16h r;
#pragma unroll
  for (int i = 0; i < 8; ++i) { r[i] = lo[i]; r[i + 8] = hi[i]; }
  return r;
}
// B-fragment (32x16 f16): lane%16 = column, K rows {k0..k0+7},{k0+16..k0+23} (+8 for lane>=16), row stride ld.
__device__ __forceinline__ v16h frag_row16(const _Float16* p, int ld) {
  v16h r;
#pragma unroll
  for (int i = 0; i < 8; ++i) { r[i] = p[(size_t)i * ld]; r[i + 8] = p[(size_t)(16 + i) * ld]; }
  return r;
}
__device__ __forceinline__ v8f wmma32(v16h a, v16h b, v8f c) {
  return __builtin_amdgcn_wmma_f32_16x16x32_f16(false, a, false, b, (short)0, c,
                                                false, false);
}

// ---------------- K0: zero the global scalar accumulator ----------------
__global__ void k_zero(float* p) {
  if (threadIdx.x == 0 && blockIdx.x == 0) *p = 0.f;
}

// ---------------- K1a: elementwise fp32 -> f16 ------------------------------
__global__ void k_convX(const float* __restrict__ src, _Float16* __restrict__ dst,
                        int total) {
  int idx = blockIdx.x * blockDim.x + threadIdx.x;
  if (idx < total) dst[idx] = (_Float16)src[idx];
}

// ---------------- K1b: convert + transpose weights fp32 -> f16 K-major ------
// dst[h][k][n] = (n < N) ? src[h][n][k] : 0     (dst Npad columns)
__global__ void k_convT(const float* __restrict__ src, _Float16* __restrict__ dst,
                        int Hn, int N, int K, int Npad) {
  int idx = blockIdx.x * blockDim.x + threadIdx.x;
  int total = Hn * K * Npad;
  if (idx >= total) return;
  int n = idx % Npad;
  int k = (idx / Npad) % K;
  int h = idx / (Npad * K);
  float v = (n < N) ? src[((size_t)h * N + n) * K + k] : 0.f;
  dst[idx] = (_Float16)v;
}

// ---------------- K2: hid = x @ W_shared^T + b_shared  (f16 in/out) ---------
__global__ __launch_bounds__(256) void k_hid(
    const _Float16* __restrict__ x16, const _Float16* __restrict__ Wst,
    const float* __restrict__ bsh, _Float16* __restrict__ hid) {
  const int lane = threadIdx.x & 31;
  const int w    = threadIdx.x >> 5;
  const int t    = blockIdx.x * 8 + w;          // 32768 wave-tiles total
  const int row0 = (t >> 5) * 16;
  const int n0   = (t & 31) * 16;
  const int r = lane & 15, hh = lane >> 4, n = r, mb = hh * 8;

  float bias = bsh[n0 + n];
  v8f acc;
#pragma unroll
  for (int i = 0; i < 8; ++i) acc[i] = bias;

#pragma unroll 1
  for (int k0 = 0; k0 < DX_; k0 += 32) {
    v16h af = frag_gap16(x16 + (size_t)(row0 + r) * DX_ + k0 + hh * 8);
    v16h bf = frag_row16(Wst + (size_t)(k0 + hh * 8) * DS_ + n0 + r, DS_);
    acc = wmma32(af, bf, acc);
  }
#pragma unroll
  for (int v = 0; v < 8; ++v)
    hid[(size_t)(row0 + mb + v) * DS_ + n0 + n] = (_Float16)acc[v];
}

// ---------------- K3: fused per-(16-row tile, head) MLP chain ---------------
__global__ __launch_bounds__(128) void k_mlp(
    const _Float16* __restrict__ hid,
    const _Float16* __restrict__ W1t, const float* __restrict__ b1,
    const _Float16* __restrict__ W2t, const float* __restrict__ b2,
    const _Float16* __restrict__ Wot, const float* __restrict__ bo,
    const _Float16* __restrict__ Wqt, const float* __restrict__ bq,
    _Float16* __restrict__ qout, float* __restrict__ meanr,
    float* __restrict__ stdr) {
  __shared__ __align__(16) _Float16 buf[4][16][256];   // 32 KB, per-wave slices
  const int lane = threadIdx.x & 31;
  const int w    = threadIdx.x >> 5;
  const int task = blockIdx.x * 4 + w;                 // 16384 tasks
  const int head = task & (NH_ - 1);
  const int row0 = (task >> 4) * 16;
  const int r = lane & 15, hh = lane >> 4, n = r, mb = hh * 8;
  _Float16(*mybuf)[256] = buf[w];

  // hid A-fragments for this 16-row tile (K = 512 -> 16 fragments)
  v16h Ah[16];
#pragma unroll
  for (int k = 0; k < 16; ++k)
    Ah[k] = frag_gap16(hid + (size_t)(row0 + r) * DS_ + k * 32 + hh * 8);

  // GEMM1: h1 = relu(hid @ W1[h]^T + b1[h])   -> LDS (f16)
  const _Float16* W1h = W1t + (size_t)head * DS_ * H1_;
#pragma unroll 1
  for (int nt = 0; nt < 16; ++nt) {
    if (nt < 15)   // pull next weight column block toward the WGP cache
      __builtin_prefetch(W1h + (size_t)lane * H1_ + (nt + 1) * 16, 0, 3);
    float bias = b1[head * H1_ + nt * 16 + n];
    v8f acc;
#pragma unroll
    for (int i = 0; i < 8; ++i) acc[i] = bias;
#pragma unroll
    for (int k = 0; k < 16; ++k) {
      v16h bf = frag_row16(W1h + (size_t)(k * 32 + hh * 8) * H1_ + nt * 16 + r, H1_);
      acc = wmma32(Ah[k], bf, acc);
    }
#pragma unroll
    for (int v = 0; v < 8; ++v)
      mybuf[mb + v][nt * 16 + n] = (_Float16)fmaxf(acc[v], 0.f);
  }
  asm volatile("s_wait_dscnt 0" ::: "memory");

  // reload h1 as A-fragments (LDS transposes D-layout -> A-layout)
  v16h A1[8];
#pragma unroll
  for (int k = 0; k < 8; ++k)
    A1[k] = frag_gap16(&mybuf[r][k * 32 + hh * 8]);

  // GEMM2: h2 = relu(h1 @ W2[h]^T + b2[h])    -> LDS (f16, overwrites h1)
  const _Float16* W2h = W2t + (size_t)head * H1_ * H2_;
#pragma unroll 1
  for (int nt = 0; nt < 16; ++nt) {
    if (nt < 15)
      __builtin_prefetch(W2h + (size_t)lane * H2_ + (nt + 1) * 16, 0, 3);
    float bias = b2[head * H2_ + nt * 16 + n];
    v8f acc;
#pragma unroll
    for (int i = 0; i < 8; ++i) acc[i] = bias;
#pragma unroll
    for (int k = 0; k < 8; ++k) {
      v16h bf = frag_row16(W2h + (size_t)(k * 32 + hh * 8) * H2_ + nt * 16 + r, H2_);
      acc = wmma32(A1[k], bf, acc);
    }
#pragma unroll
    for (int v = 0; v < 8; ++v)
      mybuf[mb + v][nt * 16 + n] = (_Float16)fmaxf(acc[v], 0.f);
  }
  asm volatile("s_wait_dscnt 0" ::: "memory");

  v16h A2[8];
#pragma unroll
  for (int k = 0; k < 8; ++k)
    A2[k] = frag_gap16(&mybuf[r][k * 32 + hh * 8]);

  // GEMM3: out = h2 @ Wo[h]^T + bo[h]  (16x16 tile, cols 0..11 valid)
  {
    float bias = (n < 12) ? bo[head * 12 + n] : 0.f;
    v8f acc;
#pragma unroll
    for (int i = 0; i < 8; ++i) acc[i] = bias;
    const _Float16* Woh = Wot + (size_t)head * H2_ * 16;
#pragma unroll
    for (int k = 0; k < 8; ++k) {
      v16h bf = frag_row16(Woh + (size_t)(k * 32 + hh * 8) * 16 + r, 16);
      acc = wmma32(A2[k], bf, acc);
    }
#pragma unroll
    for (int v = 0; v < 8; ++v) {
      int row = row0 + mb + v;
      if (n < 6)
        meanr[((size_t)row * NH_ + head) * O_ + n] = acc[v];
      else if (n < 12)
        stdr[((size_t)row * NH_ + head) * O_ + (n - 6)] = acc[v];
    }
  }

  // GEMM4: q = h2 @ Wq^T + bq  (16x128)  -> ws (f16)
#pragma unroll 1
  for (int nt = 0; nt < 8; ++nt) {
    float bias = bq[nt * 16 + n];
    v8f acc;
#pragma unroll
    for (int i = 0; i < 8; ++i) acc[i] = bias;
#pragma unroll
    for (int k = 0; k < 8; ++k) {
      v16h bf = frag_row16(Wqt + (size_t)(k * 32 + hh * 8) * A_ + nt * 16 + r, A_);
      acc = wmma32(A2[k], bf, acc);
    }
#pragma unroll
    for (int v = 0; v < 8; ++v) {
      int row = row0 + mb + v;
      qout[((size_t)row * NH_ + head) * A_ + nt * 16 + n] = (_Float16)acc[v];
    }
  }
}

// ---------------- K4: attention combine per 16-row tile ---------------------
__global__ __launch_bounds__(256) void k_attn(
    const float* __restrict__ skill, const float* __restrict__ Wk,
    const float* __restrict__ bk, const _Float16* __restrict__ qbuf,
    const float* __restrict__ meanr, const float* __restrict__ stdr,
    float* __restrict__ out, float* __restrict__ Sacc) {
  __shared__ float sk[16][16];
  __shared__ float kv[16][128];
  __shared__ float invk[16];
  __shared__ float invq[16][128];
  __shared__ float sc[16][16];
  __shared__ float red[256];
  const int t = threadIdx.x;
  const int base = blockIdx.x * 16;

  sk[t >> 4][t & 15] = skill[(size_t)(base + (t >> 4)) * NH_ + (t & 15)];
  __syncthreads();

  // kvec = skill @ Wk^T + bk
  for (int idx = t; idx < 16 * 128; idx += 256) {
    int row = idx >> 7, a = idx & 127;
    float s = bk[a];
#pragma unroll
    for (int j = 0; j < 16; ++j) s += sk[row][j] * Wk[a * 16 + j];
    kv[row][a] = s;
  }
  // 1/max(||q[b, :, a]||_h, eps)  (normalize over heads, axis=1)
  for (int idx = t; idx < 16 * 128; idx += 256) {
    int row = idx >> 7, a = idx & 127;
    float s = 0.f;
#pragma unroll
    for (int h = 0; h < 16; ++h) {
      float v = (float)qbuf[((size_t)(base + row) * NH_ + h) * A_ + a];
      s += v * v;
    }
    invq[row][a] = 1.0f / fmaxf(sqrtf(s), 1e-12f);
  }
  __syncthreads();
  if (t < 16) {
    float s = 0.f;
    for (int a = 0; a < 128; ++a) { float v = kv[t][a]; s += v * v; }
    invk[t] = 1.0f / fmaxf(sqrtf(s), 1e-12f);
  }
  __syncthreads();

  // scores[row][h] = sum_a qhat * khat
  {
    int row = t >> 4, h = t & 15;
    const _Float16* qp = qbuf + ((size_t)(base + row) * NH_ + h) * A_;
    float s = 0.f;
    for (int a = 0; a < 128; ++a)
      s += (float)qp[a] * invq[row][a] * kv[row][a];
    sc[row][h] = s * invk[row];
  }
  __syncthreads();

  float localS = 0.f;
  if (t < 16) {
    int row = t;
    float mx = -1e30f;
#pragma unroll
    for (int h = 0; h < 16; ++h) mx = fmaxf(mx, sc[row][h]);
    float e[16], sum = 0.f;
#pragma unroll
    for (int h = 0; h < 16; ++h) { e[h] = __expf(sc[row][h] - mx); sum += e[h]; }
    float isum = 1.0f / sum;
#pragma unroll
    for (int o = 0; o < O_; ++o) {
      float si = 0.f;
#pragma unroll
      for (int h = 0; h < 16; ++h) {
        float wv = e[h] * isum;
        size_t ix = ((size_t)(base + row) * NH_ + h) * O_ + o;
        float sd = stdr[ix];
        float iv = wv / (sd * sd);          // inv = w / std^2
        si += iv;
        localS += iv * meanr[ix];
      }
      float fs = 1.0f / sqrtf(si);
      out[2 * B_ * O_ + (size_t)(base + row) * O_ + o] = fs;        // final_stds
      out[1 * B_ * O_ + (size_t)(base + row) * O_ + o] = logf(fs);  // log stds
    }
  }
  red[t] = localS;
  __syncthreads();
  for (int off = 128; off > 0; off >>= 1) {
    if (t < off) red[t] += red[t + off];
    __syncthreads();
  }
  if (t == 0) atomicAdd(Sacc, red[0]);
}

// ---------------- K5: final_means = final_stds^2 * S ------------------------
__global__ void k_final(float* __restrict__ out, const float* __restrict__ Sacc) {
  int i = blockIdx.x * blockDim.x + threadIdx.x;
  if (i < B_ * O_) {
    float fs = out[2 * B_ * O_ + i];
    out[i] = fs * fs * (*Sacc);
  }
}

// ---------------- launcher --------------------------------------------------
extern "C" void kernel_launch(void* const* d_in, const int* in_sizes, int n_in,
                              void* d_out, int out_size, void* d_ws,
                              size_t ws_size, hipStream_t stream) {
  (void)in_sizes; (void)n_in; (void)out_size; (void)ws_size;
  const float* x     = (const float*)d_in[0];
  const float* skill = (const float*)d_in[1];
  const float* Wsh   = (const float*)d_in[2];
  const float* bsh   = (const float*)d_in[3];
  const float* W1    = (const float*)d_in[4];
  const float* b1    = (const float*)d_in[5];
  const float* W2    = (const float*)d_in[6];
  const float* b2    = (const float*)d_in[7];
  const float* Wo    = (const float*)d_in[8];
  const float* bo    = (const float*)d_in[9];
  const float* Wk    = (const float*)d_in[10];
  const float* bk    = (const float*)d_in[11];
  const float* Wq    = (const float*)d_in[12];
  const float* bq    = (const float*)d_in[13];

  char* ws = (char*)d_ws;
  _Float16* Wst  = (_Float16*)(ws + OFF_WST);
  _Float16* W1t  = (_Float16*)(ws + OFF_W1T);
  _Float16* W2t  = (_Float16*)(ws + OFF_W2T);
  _Float16* Wot  = (_Float16*)(ws + OFF_WOT);
  _Float16* Wqt  = (_Float16*)(ws + OFF_WQT);
  _Float16* x16  = (_Float16*)(ws + OFF_X16);
  _Float16* hid  = (_Float16*)(ws + OFF_HID);
  _Float16* qbuf = (_Float16*)(ws + OFF_Q);
  float*    mnr  = (float*)(ws + OFF_MEAN);
  float*    sdr  = (float*)(ws + OFF_STD);
  float*    S    = (float*)(ws + OFF_S);

  k_zero<<<1, 1, 0, stream>>>(S);

  k_convX<<<(B_ * DX_ + 255) / 256, 256, 0, stream>>>(x, x16, B_ * DX_);
  k_convT<<<(512 * 512 + 255) / 256, 256, 0, stream>>>(Wsh, Wst, 1, 512, 512, 512);
  k_convT<<<(16 * 512 * 256 + 255) / 256, 256, 0, stream>>>(W1, W1t, 16, 256, 512, 256);
  k_convT<<<(16 * 256 * 256 + 255) / 256, 256, 0, stream>>>(W2, W2t, 16, 256, 256, 256);
  k_convT<<<(16 * 256 * 16 + 255) / 256, 256, 0, stream>>>(Wo, Wot, 16, 12, 256, 16);
  k_convT<<<(256 * 128 + 255) / 256, 256, 0, stream>>>(Wq, Wqt, 1, 128, 256, 128);

  // 1024 row-tiles x 32 col-tiles = 32768 wave-tiles / 8 waves per block
  k_hid<<<4096, 256, 0, stream>>>(x16, Wst, bsh, hid);

  // 1024 row-tiles x 16 heads = 16384 wave-tasks / 4 waves per block
  k_mlp<<<4096, 128, 0, stream>>>(hid, W1t, b1, W2t, b2, Wot, bo, Wqt, bq,
                                  qbuf, mnr, sdr);

  k_attn<<<B_ / 16, 256, 0, stream>>>(skill, Wk, bk, qbuf, mnr, sdr,
                                      (float*)d_out, S);

  k_final<<<(B_ * O_ + 255) / 256, 256, 0, stream>>>((float*)d_out, S);
}
